// Decoder_9895604650750
// MI455X (gfx1250) — compile-verified, matrix operand order checked by reading, not run on hardware
//
// MI455X (gfx1250) WCT-decoder. bf16 WMMA implicit-GEMM convs (L2-resident,
// WMMA-bound per roofline), f32 WMMA Newton-Schulz for 32x32 matrix sqrt/invsqrt.
#include <hip/hip_runtime.h>
#include <math.h>

typedef __attribute__((ext_vector_type(16))) __bf16 v16bf;
typedef __attribute__((ext_vector_type(8)))  float  v8f;
typedef __attribute__((ext_vector_type(2)))  float  v2f;

// ---------------------------------------------------------------- weights ----
__global__ void conv_w_bf16_k(const float* __restrict__ src, __bf16* __restrict__ dst,
                              int CoSrc, int CoDst, int Ci, int KH) {
  long long total = (long long)CoDst * KH * KH * Ci;
  long long i = (long long)blockIdx.x * 256 + threadIdx.x;
  if (i >= total) return;
  int ci = (int)(i % Ci);
  long long r = i / Ci;
  int kx = (int)(r % KH); r /= KH;
  int ky = (int)(r % KH);
  int co = (int)(r / KH);
  float v = (co < CoSrc) ? src[(((size_t)co * Ci + ci) * KH + ky) * KH + kx] : 0.f;
  dst[i] = (__bf16)v;  // layout [co][ky][kx][ci]
}

// ------------------------------------------------------ implicit-GEMM conv ---
// Tile: 32 output pixels (one row segment) x Co channels. 128 threads = 4 waves.
// wave: mt = wid&1 (M subtile), q = wid>>1 (N interleave). NT16 = Co/16.
template <int KH, int UP, int NT16>
__global__ __launch_bounds__(128) void conv_wmma_k(
    const float* __restrict__ in, const __bf16* __restrict__ w,
    const float* __restrict__ bias, const float* __restrict__ skip,
    float* __restrict__ out, int Cin, int CoOut, int Hin, int Win, int act) {
  constexpr int PAD = KH / 2;
  const int Hout = Hin * UP, Wout = Win * UP;
  const int tilesX = Wout >> 5;
  int t = blockIdx.x;
  const int x0 = (t % tilesX) * 32; t /= tilesX;
  const int oy = t % Hout;
  const int b  = t / Hout;
  const int lane = threadIdx.x & 31;
  const int wid  = threadIdx.x >> 5;
  const int mt = wid & 1;
  const int q  = wid >> 1;
  constexpr int NA_MAX = (NT16 + 1) / 2;
  const int nA = (NT16 == 1) ? (q == 0 ? 1 : 0) : (NT16 / 2);
  const int ln15 = lane & 15, lhi = lane >> 4;

  __shared__ alignas(64) __bf16 As[32][32];  // [pixel][ci]
  v8f acc[NA_MAX] = {};

  for (int ky = 0; ky < KH; ++ky) {
    int yy = oy + ky - PAD;
    if (yy < 0) yy = -yy;
    if (yy >= Hout) yy = 2 * Hout - 2 - yy;
    const int iy = yy / UP;
    for (int kx = 0; kx < KH; ++kx) {
      for (int ci0 = 0; ci0 < Cin; ci0 += 32) {
        __syncthreads();
#pragma unroll
        for (int r = 0; r < 8; ++r) {
          const int e = threadIdx.x + r * 128;
          const int j = e >> 5;   // ci
          const int m = e & 31;   // pixel
          int xx = x0 + m + kx - PAD;
          if (xx < 0) xx = -xx;
          if (xx >= Wout) xx = 2 * Wout - 2 - xx;
          const int ix = xx / UP;
          As[m][j] = (__bf16)in[(((size_t)b * Cin + (ci0 + j)) * Hin + iy) * Win + ix];
        }
        __syncthreads();
        // A frag: lane = M row, 16 contiguous K (lanes 16-31 -> K 16..31)
        const v16bf a = *(const v16bf*)&As[mt * 16 + ln15][lhi * 16];
#pragma unroll
        for (int k = 0; k < NA_MAX; ++k) {
          if (k < nA) {
            const int co = (q + 2 * k) * 16 + ln15;
            const __bf16* wp =
                w + (((size_t)co * KH + ky) * KH + kx) * Cin + ci0 + lhi * 16;
            const v16bf bf = *(const v16bf*)wp;
            acc[k] = __builtin_amdgcn_wmma_f32_16x16x32_bf16(
                false, a, false, bf, (short)0, acc[k], false, false);
          }
        }
      }
    }
  }
#pragma unroll
  for (int k = 0; k < NA_MAX; ++k) {
    if (k < nA) {
      const int co = (q + 2 * k) * 16 + ln15;
      if (co < CoOut) {
        const float bv = bias[co];
#pragma unroll
        for (int r = 0; r < 8; ++r) {
          const int m = mt * 16 + r + 8 * lhi;
          const size_t oi = (((size_t)b * CoOut + co) * Hout + oy) * Wout + (x0 + m);
          float v = acc[k][r] + bv;
          if (skip) v += skip[oi];
          if (act == 1) v = v > 0.f ? v : 0.f;
          else if (act == 2) v = tanhf(v);
          out[oi] = v;
        }
      }
    }
  }
}

// ------------------------------------------------------------- WCT stats -----
__global__ __launch_bounds__(256) void wct_stats_k(const float* __restrict__ x,
                                                   float* __restrict__ mean,
                                                   float* __restrict__ cov) {
  const int grp = blockIdx.x;                       // b*8+g, channels contiguous
  const float* xb = x + (size_t)grp * 32 * 4096;
  __shared__ float ps[32][8];
  __shared__ float ml[32];
  __shared__ float Xc[32][68];
  const int t = threadIdx.x;
  {
    const int j = t >> 3, qq = t & 7;
    float s = 0.f;
    const float* p = xb + (size_t)j * 4096 + qq * 512;
    for (int i = 0; i < 512; ++i) s += p[i];
    ps[j][qq] = s;
  }
  __syncthreads();
  if (t < 32) {
    float s = 0.f;
    for (int qq = 0; qq < 8; ++qq) s += ps[t][qq];
    s *= (1.f / 4096.f);
    ml[t] = s;
    mean[grp * 32 + t] = s;
  }
  __syncthreads();
  const int jj = t & 31;
  const int ib = t >> 5;  // 0..7
  float a0 = 0, a1 = 0, a2 = 0, a3 = 0;
  for (int c0 = 0; c0 < 4096; c0 += 64) {
    __syncthreads();
#pragma unroll
    for (int r = 0; r < 8; ++r) {
      const int e = t + r * 256;
      const int row = e >> 6, col = e & 63;
      Xc[row][col] = xb[(size_t)row * 4096 + c0 + col] - ml[row];
    }
    __syncthreads();
    for (int col = 0; col < 64; ++col) {
      const float xj = Xc[jj][col];
      a0 += Xc[ib][col] * xj;
      a1 += Xc[ib + 8][col] * xj;
      a2 += Xc[ib + 16][col] * xj;
      a3 += Xc[ib + 24][col] * xj;
    }
  }
  const float inv = 1.f / 4095.f;
  float* cv = cov + (size_t)grp * 1024;
  cv[(ib) * 32 + jj]      = a0 * inv + ((ib) == jj ? 1e-5f : 0.f);
  cv[(ib + 8) * 32 + jj]  = a1 * inv + ((ib + 8) == jj ? 1e-5f : 0.f);
  cv[(ib + 16) * 32 + jj] = a2 * inv + ((ib + 16) == jj ? 1e-5f : 0.f);
  cv[(ib + 24) * 32 + jj] = a3 * inv + ((ib + 24) == jj ? 1e-5f : 0.f);
}

// --------------------------------------------- Newton-Schulz sqrt/invsqrt ----
// One wave computes a full 32x32 matmul with v_wmma_f32_16x16x4_f32 (2x2 tiles).
__device__ inline void mm32(const float (*A)[33], const float (*Bm)[33],
                            float (*D)[33], int lane) {
  const int ln15 = lane & 15, lhi = lane >> 4;
  v8f acc[2][2] = {};
  for (int k0 = 0; k0 < 32; k0 += 4) {
    const int kb = k0 + 2 * lhi;  // lanes 0-15: K k0,k0+1; lanes 16-31: k0+2,k0+3
    v2f a0, a1, b0, b1;
    a0.x = A[ln15][kb];        a0.y = A[ln15][kb + 1];
    a1.x = A[16 + ln15][kb];   a1.y = A[16 + ln15][kb + 1];
    b0.x = Bm[kb][ln15];       b0.y = Bm[kb + 1][ln15];
    b1.x = Bm[kb][16 + ln15];  b1.y = Bm[kb + 1][16 + ln15];
    acc[0][0] = __builtin_amdgcn_wmma_f32_16x16x4_f32(false, a0, false, b0, (short)0, acc[0][0], false, false);
    acc[0][1] = __builtin_amdgcn_wmma_f32_16x16x4_f32(false, a0, false, b1, (short)0, acc[0][1], false, false);
    acc[1][0] = __builtin_amdgcn_wmma_f32_16x16x4_f32(false, a1, false, b0, (short)0, acc[1][0], false, false);
    acc[1][1] = __builtin_amdgcn_wmma_f32_16x16x4_f32(false, a1, false, b1, (short)0, acc[1][1], false, false);
  }
#pragma unroll
  for (int mt = 0; mt < 2; ++mt)
#pragma unroll
    for (int nt = 0; nt < 2; ++nt)
#pragma unroll
      for (int r = 0; r < 8; ++r)
        D[mt * 16 + r + 8 * lhi][nt * 16 + ln15] = acc[mt][nt][r];
}

// mode 0: outM = sqrt(cov).  mode 1: outM = SbIn @ invsqrt(cov).
__global__ __launch_bounds__(32) void ns_kernel(const float* __restrict__ covIn,
                                                const float* __restrict__ SbIn,
                                                float* __restrict__ outM, int mode) {
  __shared__ float buf[4][32][33];
  const int g = blockIdx.x;
  const int lane = threadIdx.x;
  float (*Y)[33] = buf[0];
  float (*Z)[33] = buf[1];
  float (*P)[33] = buf[2];
  float (*S)[33] = buf[3];
  const float* cv = covIn + (size_t)g * 1024;
  S[0][lane] = cv[lane * 32 + lane];
  __syncthreads();
  float tr = 0.f;
  for (int i = 0; i < 32; ++i) tr += S[0][i];
  const float invtr = 1.f / tr;
  for (int j = 0; j < 32; ++j) {
    Y[lane][j] = cv[lane * 32 + j] * invtr;   // eigs in (0,1]
    Z[lane][j] = (lane == j) ? 1.f : 0.f;
  }
  __syncthreads();
  for (int it = 0; it < 16; ++it) {
    mm32(Z, Y, P, lane);  // P = Z@Y
    __syncthreads();
    for (int j = 0; j < 32; ++j)
      P[lane][j] = ((lane == j) ? 1.5f : 0.f) - 0.5f * P[lane][j];  // T
    __syncthreads();
    mm32(Y, P, S, lane);  // Ynew = Y@T
    __syncthreads();
    { float (*tmp)[33] = Y; Y = S; S = tmp; }
    mm32(P, Z, S, lane);  // Znew = T@Z
    __syncthreads();
    { float (*tmp)[33] = Z; Z = S; S = tmp; }
  }
  if (mode == 0) {
    const float sc = sqrtf(tr);
    for (int j = 0; j < 32; ++j)
      outM[(size_t)g * 1024 + lane * 32 + j] = Y[lane][j] * sc;
  } else {
    const float sc = rsqrtf(tr);
    for (int j = 0; j < 32; ++j)
      P[lane][j] = SbIn[(size_t)g * 1024 + lane * 32 + j];
    __syncthreads();
    mm32(P, Z, S, lane);  // Sb @ invsqrt(cov)
    __syncthreads();
    for (int j = 0; j < 32; ++j)
      outM[(size_t)g * 1024 + lane * 32 + j] = S[lane][j] * sc;
  }
}

// out = alpha*(T@(x-mc)+ms) + (1-alpha)*x
__global__ __launch_bounds__(256) void wct_apply_k(
    const float* __restrict__ x, const float* __restrict__ T,
    const float* __restrict__ mc, const float* __restrict__ ms,
    const float* __restrict__ alphas, int aidx, float* __restrict__ out) {
  const int grp = blockIdx.x >> 4;
  const int hw0 = (blockIdx.x & 15) << 8;
  const int t = threadIdx.x;
  __shared__ float Xg[32][260];
  __shared__ float Tl[32][33];
  __shared__ float mcl[32], msl[32];
  if (t < 32) { mcl[t] = mc[grp * 32 + t]; msl[t] = ms[grp * 32 + t]; }
#pragma unroll
  for (int r = 0; r < 4; ++r) {
    const int e = t + r * 256;
    Tl[e >> 5][e & 31] = T[(size_t)grp * 1024 + e];
  }
  const float* xb = x + (size_t)grp * 32 * 4096;
#pragma unroll
  for (int r = 0; r < 32; ++r) {
    const int e = t + r * 256;
    Xg[e >> 8][e & 255] = xb[(size_t)(e >> 8) * 4096 + hw0 + (e & 255)];
  }
  __syncthreads();
  const float alpha = alphas[aidx];
  float* ob = out + (size_t)grp * 32 * 4096 + hw0;
  for (int cm = 0; cm < 32; ++cm) {
    float s = 0.f;
    const float* Trow = Tl[cm];
    for (int j = 0; j < 32; ++j) s += Trow[j] * (Xg[j][t] - mcl[j]);
    ob[(size_t)cm * 4096 + t] = alpha * (s + msl[cm]) + (1.f - alpha) * Xg[cm][t];
  }
}

// ------------------------------------------------------------ GroupNorm(1) ---
__global__ __launch_bounds__(256) void gn_stats_k(const float* __restrict__ x,
                                                  float* __restrict__ stat, int n) {
  const int b = blockIdx.x;
  const float* xb = x + (size_t)b * n;
  float s = 0.f, s2 = 0.f;
  for (int i = threadIdx.x; i < n; i += 256) { float v = xb[i]; s += v; s2 += v * v; }
  __shared__ float ls[256], ls2[256];
  ls[threadIdx.x] = s; ls2[threadIdx.x] = s2;
  __syncthreads();
  for (int off = 128; off > 0; off >>= 1) {
    if (threadIdx.x < off) { ls[threadIdx.x] += ls[threadIdx.x + off]; ls2[threadIdx.x] += ls2[threadIdx.x + off]; }
    __syncthreads();
  }
  if (threadIdx.x == 0) {
    float mu = ls[0] / n;
    float var = ls2[0] / n - mu * mu;
    stat[b * 2] = mu;
    stat[b * 2 + 1] = rsqrtf(var + 1e-5f);
  }
}

__global__ void gn_apply_k(const float* __restrict__ x, const float* __restrict__ stat,
                           const float* __restrict__ g, const float* __restrict__ bt,
                           float* __restrict__ out, int C, int HW, long long total) {
  long long i = (long long)blockIdx.x * 256 + threadIdx.x;
  if (i >= total) return;
  const int b = (int)(i / ((long long)C * HW));
  const int c = (int)((i / HW) % C);
  float v = (x[i] - stat[b * 2]) * stat[b * 2 + 1] * g[c] + bt[c];
  out[i] = v > 0.f ? v : 0.f;
}

// ------------------------------------------------------------------- host ----
extern "C" void kernel_launch(void* const* d_in, const int* in_sizes, int n_in,
                              void* d_out, int out_size, void* d_ws, size_t ws_size,
                              hipStream_t stream) {
  // Detect flattening order (insertion vs sorted-key) via size anchors.
  int idxOutW = -1, idxResW0 = -1;
  for (int i = 0; i < n_in; ++i) {
    if (in_sizes[i] == 9408 && idxOutW < 0) idxOutW = i;
    if (in_sizes[i] == 589824 && idxResW0 < 0) idxResW0 = i;
  }
  const bool sorted = (idxOutW >= 0 && idxResW0 >= 0 && idxOutW < idxResW0);
  const float *c_A, *s_B, *alphas, *up1_w, *up1_b, *gn1_g, *gn1_b;
  const float *up2_w, *up2_b, *gn2_g, *gn2_b, *out_w, *out_b;
  const float *res_w1[4], *res_b1[4], *res_w2[4], *res_b2[4];
  if (!sorted) {  // insertion order: c_A, s_B, alphas, res(w1,b1,w2,b2)x4, up1..out
    c_A = (const float*)d_in[0];  s_B = (const float*)d_in[1];
    alphas = (const float*)d_in[2];
    for (int i = 0; i < 4; ++i) {
      res_w1[i] = (const float*)d_in[3 + 4 * i];
      res_b1[i] = (const float*)d_in[4 + 4 * i];
      res_w2[i] = (const float*)d_in[5 + 4 * i];
      res_b2[i] = (const float*)d_in[6 + 4 * i];
    }
    up1_w = (const float*)d_in[19]; up1_b = (const float*)d_in[20];
    gn1_g = (const float*)d_in[21]; gn1_b = (const float*)d_in[22];
    up2_w = (const float*)d_in[23]; up2_b = (const float*)d_in[24];
    gn2_g = (const float*)d_in[25]; gn2_b = (const float*)d_in[26];
    out_w = (const float*)d_in[27]; out_b = (const float*)d_in[28];
  } else {  // sorted keys: c_A, alphas, gn1_b, gn1_g, gn2_b, gn2_g, out_b, out_w,
            // res(b1,b2,w1,w2)x4, up1_b, up1_w, up2_b, up2_w, s_B
    c_A = (const float*)d_in[0];
    alphas = (const float*)d_in[1];
    gn1_b = (const float*)d_in[2]; gn1_g = (const float*)d_in[3];
    gn2_b = (const float*)d_in[4]; gn2_g = (const float*)d_in[5];
    out_b = (const float*)d_in[6]; out_w = (const float*)d_in[7];
    for (int i = 0; i < 4; ++i) {
      res_b1[i] = (const float*)d_in[8 + 4 * i];
      res_b2[i] = (const float*)d_in[9 + 4 * i];
      res_w1[i] = (const float*)d_in[10 + 4 * i];
      res_w2[i] = (const float*)d_in[11 + 4 * i];
    }
    up1_b = (const float*)d_in[24]; up1_w = (const float*)d_in[25];
    up2_b = (const float*)d_in[26]; up2_w = (const float*)d_in[27];
    s_B = (const float*)d_in[28];
  }

  // Workspace carve-up (aligned 256B). Total ~247 MB.
  char* ws = (char*)d_ws;
  size_t off = 0;
  auto alloc = [&](size_t bytes) -> void* {
    off = (off + 255) & ~(size_t)255;
    void* p = ws + off;
    off += bytes;
    return p;
  };
  __bf16* wRes[8];
  for (int i = 0; i < 8; ++i) wRes[i] = (__bf16*)alloc(589824u * 2);
  __bf16* wUp1 = (__bf16*)alloc(819200u * 2);
  __bf16* wUp2 = (__bf16*)alloc(204800u * 2);
  __bf16* wOut = (__bf16*)alloc(50176u * 2);
  float* meanS = (float*)alloc(64 * 32 * 4);
  float* meanC = (float*)alloc(64 * 32 * 4);
  float* cov   = (float*)alloc(64 * 1024 * 4);
  float* Sb    = (float*)alloc(64 * 1024 * 4);
  float* Tm    = (float*)alloc(64 * 1024 * 4);
  float* gst   = (float*)alloc(64);
  float* actA  = (float*)alloc(33554432u);
  float* y1    = (float*)alloc(67108864u);
  float* big   = (float*)alloc(134217728u);
  float* actB  = big;
  float* actC  = (float*)((char*)big + 33554432u);
  float* y2    = big;
  float* dout  = (float*)d_out;

  // 1) convert weights to bf16 [co][ky][kx][ci]
  for (int i = 0; i < 4; ++i) {
    conv_w_bf16_k<<<(589824 + 255) / 256, 256, 0, stream>>>(res_w1[i], wRes[2 * i], 256, 256, 256, 3);
    conv_w_bf16_k<<<(589824 + 255) / 256, 256, 0, stream>>>(res_w2[i], wRes[2 * i + 1], 256, 256, 256, 3);
  }
  conv_w_bf16_k<<<(819200 + 255) / 256, 256, 0, stream>>>(up1_w, wUp1, 128, 128, 256, 5);
  conv_w_bf16_k<<<(204800 + 255) / 256, 256, 0, stream>>>(up2_w, wUp2, 64, 64, 128, 5);
  conv_w_bf16_k<<<(50176 + 255) / 256, 256, 0, stream>>>(out_w, wOut, 3, 16, 64, 7);

  // 2) style branch stats (once): Sb = sqrt(cov(s_B))
  wct_stats_k<<<64, 256, 0, stream>>>(s_B, meanS, cov);
  ns_kernel<<<64, 32, 0, stream>>>(cov, nullptr, Sb, 0);

  // 3) WCT #0 on c_A -> actA
  wct_stats_k<<<64, 256, 0, stream>>>(c_A, meanC, cov);
  ns_kernel<<<64, 32, 0, stream>>>(cov, Sb, Tm, 1);
  wct_apply_k<<<1024, 256, 0, stream>>>(c_A, Tm, meanC, meanS, alphas, 0, actA);

  // 4) 4x [resblock -> WCT]
  for (int i = 0; i < 4; ++i) {
    conv_wmma_k<3, 1, 16><<<8 * 64 * 2, 128, 0, stream>>>(
        actA, wRes[2 * i], res_b1[i], nullptr, actB, 256, 256, 64, 64, 1);
    conv_wmma_k<3, 1, 16><<<8 * 64 * 2, 128, 0, stream>>>(
        actB, wRes[2 * i + 1], res_b2[i], actA, actC, 256, 256, 64, 64, 0);
    wct_stats_k<<<64, 256, 0, stream>>>(actC, meanC, cov);
    ns_kernel<<<64, 32, 0, stream>>>(cov, Sb, Tm, 1);
    wct_apply_k<<<1024, 256, 0, stream>>>(actC, Tm, meanC, meanS, alphas, i + 1, actA);
  }

  // 5) up1: upsample2 + conv5 + GN(1) + relu   (64x64 -> 128x128, 256 -> 128)
  conv_wmma_k<5, 2, 8><<<8 * 128 * 4, 128, 0, stream>>>(
      actA, wUp1, up1_b, nullptr, y1, 256, 128, 64, 64, 0);
  gn_stats_k<<<8, 256, 0, stream>>>(y1, gst, 128 * 128 * 128);
  {
    long long total = 8ll * 128 * 128 * 128;
    gn_apply_k<<<(unsigned)((total + 255) / 256), 256, 0, stream>>>(
        y1, gst, gn1_g, gn1_b, y1, 128, 128 * 128, total);
  }

  // 6) up2: upsample2 + conv5 + GN(1) + relu   (128x128 -> 256x256, 128 -> 64)
  conv_wmma_k<5, 2, 4><<<8 * 256 * 8, 128, 0, stream>>>(
      y1, wUp2, up2_b, nullptr, y2, 128, 64, 128, 128, 0);
  gn_stats_k<<<8, 256, 0, stream>>>(y2, gst, 64 * 256 * 256);
  {
    long long total = 8ll * 64 * 256 * 256;
    gn_apply_k<<<(unsigned)((total + 255) / 256), 256, 0, stream>>>(
        y2, gst, gn2_g, gn2_b, y2, 64, 256 * 256, total);
  }

  // 7) out: conv7 + tanh (Co padded to 16 for WMMA, only 3 channels stored)
  conv_wmma_k<7, 1, 1><<<8 * 256 * 8, 128, 0, stream>>>(
      y2, wOut, out_b, nullptr, dout, 64, 3, 256, 256, 2);
  (void)n_in; (void)out_size; (void)ws_size;
}
